// MatchModule_44547400794507
// MI455X (gfx1250) — matrix operand order repaired and checked by reading, not compiled, not run on hardware
//
#include <hip/hip_runtime.h>
#include <stdint.h>

// ---------------------------------------------------------------------------
// Types
// ---------------------------------------------------------------------------
typedef __attribute__((ext_vector_type(16))) __bf16    v16bf;
typedef __attribute__((ext_vector_type(8)))  float     v8f;
typedef __attribute__((ext_vector_type(4)))  unsigned  u32x4;
typedef __attribute__((ext_vector_type(4)))  int       i32x4;
typedef unsigned short u16;

#if defined(__gfx1250__) && __has_builtin(__builtin_amdgcn_global_load_async_to_lds_b128)
#define USE_ASYNC_LDS 1
#else
#define USE_ASYNC_LDS 0
#endif

#if USE_ASYNC_LDS
typedef __attribute__((address_space(1))) i32x4* gvec_ptr;   // global v4i32*
typedef __attribute__((address_space(3))) i32x4* lvec_ptr;   // LDS    v4i32*
#endif

union Frag {        // 8 VGPRs = 16 bf16 per lane
    u32x4 q[2];
    v16bf v;
};

__device__ __forceinline__ u16 f2bf(float f) {
    union { float f; unsigned u; } c{f};
    unsigned u = c.u;
    unsigned r = u + 0x7FFFu + ((u >> 16) & 1u);   // round-to-nearest-even
    return (u16)(r >> 16);
}

__device__ __forceinline__ void wait_async0() {
#if USE_ASYNC_LDS
#if __has_builtin(__builtin_amdgcn_s_wait_asynccnt)
    __builtin_amdgcn_s_wait_asynccnt(0);
#else
    asm volatile("s_wait_asynccnt 0x0" ::: "memory");
#endif
#endif
}

// ---------------------------------------------------------------------------
// Elementwise fp32 -> bf16 conversion
// ---------------------------------------------------------------------------
__global__ __launch_bounds__(256)
void cvt_f32_to_bf16(const float* __restrict__ in, u16* __restrict__ out, long n) {
    long i = (long)blockIdx.x * blockDim.x + threadIdx.x;
    const long stride = (long)gridDim.x * blockDim.x;
    for (; i < n; i += stride) out[i] = f2bf(in[i]);
}

// ---------------------------------------------------------------------------
// Per-batch tiled transpose + convert: out[b][c][r] = bf16(in[b][r][c])
// in: [rows][cols] fp32, out: [cols][rows] bf16. grid = (cols/32, rows/32, B)
// ---------------------------------------------------------------------------
__global__ __launch_bounds__(256)
void transpose_cvt(const float* __restrict__ in, u16* __restrict__ out,
                   int rows, int cols) {
    __shared__ u16 tile[32][34];
    const long sIn  = (long)rows * cols;
    in  += (long)blockIdx.z * sIn;
    out += (long)blockIdx.z * sIn;
    const int r0 = blockIdx.y * 32, c0 = blockIdx.x * 32;
    const int tx = threadIdx.x & 31, ty = threadIdx.x >> 5;
    #pragma unroll
    for (int i = 0; i < 32; i += 8)
        tile[ty + i][tx] = f2bf(in[(long)(r0 + ty + i) * cols + c0 + tx]);
    __syncthreads();
    #pragma unroll
    for (int i = 0; i < 32; i += 8)
        out[(long)(c0 + ty + i) * rows + r0 + tx] = tile[tx][ty + i];
}

// ---------------------------------------------------------------------------
// Generic bf16 WMMA GEMM:  C[M,N] = A[M,K] x Bt[N,K]^T  (+bias) (+relu)
// Block: 256 threads (8 waves, wave grid 4x2). Tile 128x128, K-step 32.
// LDS double buffered; tiles staged with GLOBAL_LOAD_ASYNC_TO_LDS_B128
// (ASYNCcnt) when available, classic b128 load/store otherwise.
// grid = (N/128, M/128, batch)
// ---------------------------------------------------------------------------
template<bool F32OUT, bool RELU, bool HASBIAS>
__global__ __launch_bounds__(256)
void gemm_bf16_wmma(const u16* __restrict__ A, const u16* __restrict__ Bt,
                    void* __restrict__ Cout, const float* __restrict__ bias,
                    long batchStrideA, long batchStrideB, long batchStrideC,
                    int N, int K) {
    constexpr int PITCH = 40;                     // halfs per LDS row (16B aligned)
    __shared__ u16 sm[2][2][128 * PITCH];         // [buf][A|Bt][...]  40 KB

    A  += (long)blockIdx.z * batchStrideA;
    Bt += (long)blockIdx.z * batchStrideB;

    const int m0 = blockIdx.y * 128;
    const int n0 = blockIdx.x * 128;

    const int tid  = threadIdx.x;
    const int lane = tid & 31;
    const int wid  = tid >> 5;
    const int wm   = wid >> 1;       // 0..3 : 32 rows each
    const int wn   = wid & 1;        // 0..1 : 64 cols each
    const int lm   = lane & 15;
    const int hl   = lane >> 4;      // half-wave select

    // global->LDS staging: 512 chunks of 8 halfs (16B) per tile, 2 per thread
    const int r0c = tid >> 2;
    const int c0c = (tid & 3) * 8;
    const int r1c = r0c + 64;

#if USE_ASYNC_LDS
    auto issue_tile = [&](int kpos, int b) {
        __builtin_amdgcn_global_load_async_to_lds_b128(
            (gvec_ptr)(A + (long)(m0 + r0c) * K + kpos + c0c),
            (lvec_ptr)&sm[b][0][r0c * PITCH + c0c], 0, 0);
        __builtin_amdgcn_global_load_async_to_lds_b128(
            (gvec_ptr)(A + (long)(m0 + r1c) * K + kpos + c0c),
            (lvec_ptr)&sm[b][0][r1c * PITCH + c0c], 0, 0);
        __builtin_amdgcn_global_load_async_to_lds_b128(
            (gvec_ptr)(Bt + (long)(n0 + r0c) * K + kpos + c0c),
            (lvec_ptr)&sm[b][1][r0c * PITCH + c0c], 0, 0);
        __builtin_amdgcn_global_load_async_to_lds_b128(
            (gvec_ptr)(Bt + (long)(n0 + r1c) * K + kpos + c0c),
            (lvec_ptr)&sm[b][1][r1c * PITCH + c0c], 0, 0);
    };
#else
    u32x4 pfa0, pfa1, pfb0, pfb1;
    auto fetch = [&](int kpos) {
        pfa0 = *(const u32x4*)(A  + (long)(m0 + r0c) * K + kpos + c0c);
        pfa1 = *(const u32x4*)(A  + (long)(m0 + r1c) * K + kpos + c0c);
        pfb0 = *(const u32x4*)(Bt + (long)(n0 + r0c) * K + kpos + c0c);
        pfb1 = *(const u32x4*)(Bt + (long)(n0 + r1c) * K + kpos + c0c);
    };
    auto stash = [&](int b) {
        *(u32x4*)(&sm[b][0][r0c * PITCH + c0c]) = pfa0;
        *(u32x4*)(&sm[b][0][r1c * PITCH + c0c]) = pfa1;
        *(u32x4*)(&sm[b][1][r0c * PITCH + c0c]) = pfb0;
        *(u32x4*)(&sm[b][1][r1c * PITCH + c0c]) = pfb1;
    };
#endif

    v8f acc[2][4] = {};

#if USE_ASYNC_LDS
    issue_tile(0, 0);
    wait_async0();
    __syncthreads();
#else
    fetch(0);
    stash(0);
    __syncthreads();
#endif

    int buf = 0;
    for (int k0 = 0; k0 < K; k0 += 32) {
        const bool has = (k0 + 32) < K;
#if USE_ASYNC_LDS
        if (has) issue_tile(k0 + 32, buf ^ 1);   // DMA overlaps the WMMA phase
#else
        if (has) fetch(k0 + 32);
#endif

        const u16* sa = &sm[buf][0][0];
        const u16* sb = &sm[buf][1][0];

        // A 16x32 bf16 frag: lane(<16): v0-3 = K0..7, v4-7 = K16..23;
        //                    lane(>=16): K8..15 / K24..31
        Frag af[2];
        #pragma unroll
        for (int mt = 0; mt < 2; ++mt) {
            const int m = wm * 32 + mt * 16 + lm;
            af[mt].q[0] = *(const u32x4*)(sa + m * PITCH + hl * 8);
            af[mt].q[1] = *(const u32x4*)(sa + m * PITCH + hl * 8 + 16);
        }
        // B 32x16 bf16 frag (from Bt rows): lane(<16): K0..7 / K16..23, lane(>=16): K8..15 / K24..31
        Frag bfr[4];
        #pragma unroll
        for (int nt = 0; nt < 4; ++nt) {
            const int n = wn * 64 + nt * 16 + lm;
            bfr[nt].q[0] = *(const u32x4*)(sb + n * PITCH + hl * 8);
            bfr[nt].q[1] = *(const u32x4*)(sb + n * PITCH + hl * 8 + 16);
        }

        #pragma unroll
        for (int mt = 0; mt < 2; ++mt)
            #pragma unroll
            for (int nt = 0; nt < 4; ++nt)
                acc[mt][nt] = __builtin_amdgcn_wmma_f32_16x16x32_bf16(
                    false, af[mt].v, false, bfr[nt].v,
                    (short)0, acc[mt][nt], false, false);

#if USE_ASYNC_LDS
        wait_async0();            // my async copies into buf^1 are complete
#else
        if (has) stash(buf ^ 1);
#endif
        __syncthreads();
        buf ^= 1;
    }

    // Epilogue: C/D layout -> row = v + 8*hl, col = lm (per 16x16 tile)
    #pragma unroll
    for (int mt = 0; mt < 2; ++mt) {
        #pragma unroll
        for (int nt = 0; nt < 4; ++nt) {
            const int nn = n0 + wn * 64 + nt * 16 + lm;
            const int mb = m0 + wm * 32 + mt * 16 + hl * 8;
            float bv = 0.0f;
            if (HASBIAS) bv = bias[nn];
            #pragma unroll
            for (int v = 0; v < 8; ++v) {
                float x = acc[mt][nt][v] + bv;
                if (RELU) x = fmaxf(x, 0.0f);
                const long idx = (long)blockIdx.z * batchStrideC
                               + (long)(mb + v) * N + nn;
                if (F32OUT) ((float*)Cout)[idx] = x;
                else        ((u16*)Cout)[idx]   = f2bf(x);
            }
        }
    }
}

// ---------------------------------------------------------------------------
// Row softmax: one 256-thread block per row of length L=2048 (fp32 in, bf16 out)
// ---------------------------------------------------------------------------
__global__ __launch_bounds__(256)
void softmax_rows(const float* __restrict__ S, u16* __restrict__ P, int L) {
    __shared__ float red[256];
    const long row = blockIdx.x;
    const float* s = S + row * (long)L;
    u16* p = P + row * (long)L;
    const int tid = threadIdx.x;

    float v[8];
    float mx = -3.4e38f;
    #pragma unroll
    for (int i = 0; i < 8; ++i) {
        v[i] = s[tid + i * 256];
        mx = fmaxf(mx, v[i]);
    }
    red[tid] = mx;
    __syncthreads();
    for (int o = 128; o > 0; o >>= 1) {
        if (tid < o) red[tid] = fmaxf(red[tid], red[tid + o]);
        __syncthreads();
    }
    mx = red[0];
    __syncthreads();

    float sum = 0.0f;
    #pragma unroll
    for (int i = 0; i < 8; ++i) {
        v[i] = __expf(v[i] - mx);
        sum += v[i];
    }
    red[tid] = sum;
    __syncthreads();
    for (int o = 128; o > 0; o >>= 1) {
        if (tid < o) red[tid] += red[tid + o];
        __syncthreads();
    }
    const float inv = 1.0f / red[0];
    #pragma unroll
    for (int i = 0; i < 8; ++i)
        p[tid + i * 256] = f2bf(v[i] * inv);
}

// ---------------------------------------------------------------------------
// Orchestration
// ---------------------------------------------------------------------------
extern "C" void kernel_launch(void* const* d_in, const int* in_sizes, int n_in,
                              void* d_out, int out_size, void* d_ws, size_t ws_size,
                              hipStream_t stream) {
    constexpr int B = 16, LP = 2048, LQ = 2048, H = 1024;
    const float* proj_p = (const float*)d_in[0];
    const float* proj_q = (const float*)d_in[1];
    const float* W      = (const float*)d_in[2];
    const float* bias   = (const float*)d_in[3];

    char* ws = (char*)d_ws;
    size_t off = 0;
    auto carve = [&](size_t bytes) -> void* {
        void* p = ws + off;
        off = (off + bytes + 255) & ~(size_t)255;
        return p;
    };
    u16*   Wb  = (u16*)carve((size_t)H * H * 2);           //   2 MB
    u16*   pb  = (u16*)carve((size_t)B * LP * H * 2);      //  64 MB
    u16*   qb  = (u16*)carve((size_t)B * LQ * H * 2);      //  64 MB
    u16*   qbT = (u16*)carve((size_t)B * H * LQ * 2);      //  64 MB
    u16*   tq  = (u16*)carve((size_t)B * LQ * H * 2);      //  64 MB
    u16*   av  = (u16*)carve((size_t)B * LP * H * 2);      //  64 MB
    u16*   P   = (u16*)carve((size_t)B * LP * LQ * 2);     // 128 MB
    float* S   = (float*)carve((size_t)B * LP * LQ * 4);   // 256 MB

    // bf16 conversions
    cvt_f32_to_bf16<<<4096, 256, 0, stream>>>(W, Wb, (long)H * H);
    cvt_f32_to_bf16<<<16384, 256, 0, stream>>>(proj_p, pb, (long)B * LP * H);
    cvt_f32_to_bf16<<<16384, 256, 0, stream>>>(proj_q, qb, (long)B * LQ * H);
    // qbT[b][h][q] = bf16(proj_q[b][q][h])
    transpose_cvt<<<dim3(H / 32, LQ / 32, B), 256, 0, stream>>>(proj_q, qbT, LQ, H);

    // 1) tq = qb @ Wb^T + bias          [B,LQ,H] bf16
    gemm_bf16_wmma<false, false, true><<<dim3(H / 128, LQ / 128, B), 256, 0, stream>>>(
        qb, Wb, tq, bias, (long)LQ * H, 0, (long)LQ * H, H, H);

    // 2a) S = pb @ tq^T                 [B,LP,LQ] fp32
    gemm_bf16_wmma<true, false, false><<<dim3(LQ / 128, LP / 128, B), 256, 0, stream>>>(
        pb, tq, S, nullptr, (long)LP * H, (long)LQ * H, (long)LP * LQ, LQ, H);

    // 2b) P = softmax(S, axis=-1)       [B,LP,LQ] bf16
    softmax_rows<<<B * LP, 256, 0, stream>>>(S, P, LQ);

    // 2c) av = P @ qbT^T                [B,LP,H] bf16
    gemm_bf16_wmma<false, false, false><<<dim3(H / 128, LP / 128, B), 256, 0, stream>>>(
        P, qbT, av, nullptr, (long)LP * LQ, (long)H * LQ, (long)LP * H, H, LQ);

    // 3) out = relu(av @ Wb^T + bias)   [B,LP,H] fp32
    gemm_bf16_wmma<true, true, true><<<dim3(H / 128, LP / 128, B), 256, 0, stream>>>(
        av, Wb, (float*)d_out, bias, (long)LP * H, 0, (long)LP * H, H, H);
}
